// OrkyStateSpaceModel_19258633356046
// MI455X (gfx1250) — matrix operand order repaired and checked by reading, not compiled, not run on hardware
//
#include <hip/hip_runtime.h>
#include <stdint.h>

#define DMODEL 1024
#define INNER  2048
#define NSTATE 16
#define BATCH  2
#define SEQ    1024
#define MROWS  (BATCH * SEQ)   // 2048
#define NCHUNK 16
#define LCHUNK (SEQ / NCHUNK)  // 64
#define LSTRIDE 40             // padded LDS column stride (elements); 80B, 16B-aligned

typedef __bf16 bf16_t;
typedef bf16_t v8bf  __attribute__((ext_vector_type(8)));
typedef bf16_t v16bf __attribute__((ext_vector_type(16)));
typedef float  v8f   __attribute__((ext_vector_type(8)));

union V16U { v16bf v; v8bf h[2]; };

// ---------- scalar helpers ----------
__device__ __forceinline__ uint16_t f2bf(float f) {
    uint32_t u = __float_as_uint(f);
    u += 0x7FFFu + ((u >> 16) & 1u);          // round-to-nearest-even
    return (uint16_t)(u >> 16);
}
__device__ __forceinline__ float bf2f(uint16_t h) {
    return __uint_as_float(((uint32_t)h) << 16);
}
__device__ __forceinline__ float sigmoidf_(float x) { return 1.0f / (1.0f + __expf(-x)); }
__device__ __forceinline__ float siluf_(float x)    { return x * sigmoidf_(x); }
__device__ __forceinline__ float softplusf_(float x){ return (x > 20.0f) ? x : __logf(1.0f + __expf(x)); }

// ---------- precision-split conversions ----------
// fp32 weight [K,N] row-major  ->  bf16 hi/lo, TRANSPOSED to [N,K] (WMMA-B friendly)
__global__ void k_convert_weightT(const float* __restrict__ W,
                                  uint16_t* __restrict__ hiT, uint16_t* __restrict__ loT,
                                  int K, int N) {
    size_t tid = (size_t)blockIdx.x * blockDim.x + threadIdx.x;
    size_t total = (size_t)K * N;
    if (tid >= total) return;
    int n = (int)(tid % N);
    int k = (int)(tid / N);
    float v = W[tid];
    uint16_t hi = f2bf(v);
    float lo = v - bf2f(hi);
    size_t o = (size_t)n * K + k;
    hiT[o] = hi;
    loT[o] = f2bf(lo);
}

// fp32 [M,K] row-major -> bf16 hi/lo, same layout (WMMA-A side)
__global__ void k_convert_rowmajor(const float* __restrict__ X,
                                   uint16_t* __restrict__ hi, uint16_t* __restrict__ lo,
                                   size_t total) {
    size_t tid = (size_t)blockIdx.x * blockDim.x + threadIdx.x;
    if (tid >= total) return;
    float v = X[tid];
    uint16_t h = f2bf(v);
    hi[tid] = h;
    lo[tid] = f2bf(v - bf2f(h));
}

// ---------- WMMA fragment loads ----------
// A (16-bit, 16x32): lane<16 holds K 0..15; lane>=16 holds K 16..31
__device__ __forceinline__ v16bf ldA(const uint16_t* p) {
    V16U u;
    u.h[0] = *reinterpret_cast<const v8bf*>(p);
    u.h[1] = *reinterpret_cast<const v8bf*>(p + 8);
    return u.v;
}
// B (16-bit, 32x16): lane<16 holds K 0..15 of column n; lane>=16 holds K 16..31.
// 16 contiguous elements (from [N,K]-transposed global or from the LDS slab).
__device__ __forceinline__ v16bf ldB16(const uint16_t* p) {
    V16U u;
    u.h[0] = *reinterpret_cast<const v8bf*>(p);
    u.h[1] = *reinterpret_cast<const v8bf*>(p + 8);
    return u.v;
}

// ---------- bf16x3 GEMM with LDS-staged, double-buffered B ----------
// Block: 4 waves; block tile 128(M) x 64(N); per-wave tile 32 x 64.
// MODE 0: split columns at INNER -> out0 = inner (raw), out1 = silu(gate)
// MODE 1: out0 = softplus(v + bias[col])
// MODE 2: out0 = v
template <int MODE>
__global__ void __launch_bounds__(128)
k_gemm_bf16x3(const uint16_t* __restrict__ AHi, const uint16_t* __restrict__ ALo,
              const uint16_t* __restrict__ BTHi, const uint16_t* __restrict__ BTLo,
              int M, int N, int K,
              float* __restrict__ out0, float* __restrict__ out1,
              const float* __restrict__ bias) {
    __shared__ uint16_t lHi[2][64 * LSTRIDE];
    __shared__ uint16_t lLo[2][64 * LSTRIDE];

    const int tid    = threadIdx.x;
    const int lane   = tid & 31;
    const int wave   = tid >> 5;
    const int mBase  = blockIdx.y * 128 + wave * 32;
    const int nBase  = blockIdx.x * 64;
    const int l15    = lane & 15;
    const int hiHalf = lane >> 4;                 // 0 or 1
    const int ka     = hiHalf * 16;
    const int kb     = hiHalf * 16;

    // cooperative B staging: thread -> (column, 16-element half of the 32-K chunk)
    const int sCol = tid >> 1;                    // 0..63
    const int sOff = (tid & 1) * 16;              // 0 or 16
    const uint16_t* gBHi = BTHi + (size_t)(nBase + sCol) * K + sOff;
    const uint16_t* gBLo = BTLo + (size_t)(nBase + sCol) * K + sOff;
    const int sDst = sCol * LSTRIDE + sOff;

    const uint16_t* aHiRow0 = AHi + (size_t)(mBase + l15) * K;
    const uint16_t* aHiRow1 = AHi + (size_t)(mBase + 16 + l15) * K;
    const uint16_t* aLoRow0 = ALo + (size_t)(mBase + l15) * K;
    const uint16_t* aLoRow1 = ALo + (size_t)(mBase + 16 + l15) * K;

    v8f acc[2][4];
#pragma unroll
    for (int m = 0; m < 2; ++m)
#pragma unroll
        for (int t = 0; t < 4; ++t)
#pragma unroll
            for (int r = 0; r < 8; ++r) acc[m][t][r] = 0.0f;

    const int nch = K / 32;

    // prologue: stage chunk 0
    {
        v8bf h0 = *reinterpret_cast<const v8bf*>(gBHi);
        v8bf h1 = *reinterpret_cast<const v8bf*>(gBHi + 8);
        v8bf l0 = *reinterpret_cast<const v8bf*>(gBLo);
        v8bf l1 = *reinterpret_cast<const v8bf*>(gBLo + 8);
        *reinterpret_cast<v8bf*>(&lHi[0][sDst])     = h0;
        *reinterpret_cast<v8bf*>(&lHi[0][sDst + 8]) = h1;
        *reinterpret_cast<v8bf*>(&lLo[0][sDst])     = l0;
        *reinterpret_cast<v8bf*>(&lLo[0][sDst + 8]) = l1;
    }
    __syncthreads();

    for (int c = 0; c < nch; ++c) {
        const int p  = c & 1;
        const int kc = c * 32;

        // prefetch next B chunk into registers (overlaps with WMMAs below)
        v8bf nh0, nh1, nl0, nl1;
        const bool more = (c + 1) < nch;
        if (more) {
            nh0 = *reinterpret_cast<const v8bf*>(gBHi + kc + 32);
            nh1 = *reinterpret_cast<const v8bf*>(gBHi + kc + 40);
            nl0 = *reinterpret_cast<const v8bf*>(gBLo + kc + 32);
            nl1 = *reinterpret_cast<const v8bf*>(gBLo + kc + 40);
        }

        // A fragments for this chunk (two 16-row subtiles, hi+lo)
        v16bf a0h = ldA(aHiRow0 + kc + ka);
        v16bf a1h = ldA(aHiRow1 + kc + ka);
        v16bf a0l = ldA(aLoRow0 + kc + ka);
        v16bf a1l = ldA(aLoRow1 + kc + ka);

#pragma unroll
        for (int t = 0; t < 4; ++t) {
            const int bcol = (t * 16 + l15) * LSTRIDE + kb;
            v16bf bHi = ldB16(&lHi[p][bcol]);
            v16bf bLo = ldB16(&lLo[p][bcol]);
            acc[0][t] = __builtin_amdgcn_wmma_f32_16x16x32_bf16(false, a0h, false, bHi,
                                                                (short)0, acc[0][t], false, false);
            acc[1][t] = __builtin_amdgcn_wmma_f32_16x16x32_bf16(false, a1h, false, bHi,
                                                                (short)0, acc[1][t], false, false);
            acc[0][t] = __builtin_amdgcn_wmma_f32_16x16x32_bf16(false, a0l, false, bHi,
                                                                (short)0, acc[0][t], false, false);
            acc[1][t] = __builtin_amdgcn_wmma_f32_16x16x32_bf16(false, a1l, false, bHi,
                                                                (short)0, acc[1][t], false, false);
            acc[0][t] = __builtin_amdgcn_wmma_f32_16x16x32_bf16(false, a0h, false, bLo,
                                                                (short)0, acc[0][t], false, false);
            acc[1][t] = __builtin_amdgcn_wmma_f32_16x16x32_bf16(false, a1h, false, bLo,
                                                                (short)0, acc[1][t], false, false);
        }

        if (more) {
            *reinterpret_cast<v8bf*>(&lHi[1 - p][sDst])     = nh0;
            *reinterpret_cast<v8bf*>(&lHi[1 - p][sDst + 8]) = nh1;
            *reinterpret_cast<v8bf*>(&lLo[1 - p][sDst])     = nl0;
            *reinterpret_cast<v8bf*>(&lLo[1 - p][sDst + 8]) = nl1;
            __syncthreads();
        }
    }

    // C/D layout: VGPR r -> row mSub + r + 8*hiHalf, col nBase + t*16 + (lane&15)
#pragma unroll
    for (int m = 0; m < 2; ++m) {
        const int rBase = mBase + m * 16 + hiHalf * 8;
#pragma unroll
        for (int t = 0; t < 4; ++t) {
            const int col = nBase + t * 16 + l15;
#pragma unroll
            for (int r = 0; r < 8; ++r) {
                const int row = rBase + r;
                float v = acc[m][t][r];
                if (MODE == 0) {
                    if (col < INNER) out0[(size_t)row * INNER + col] = v;
                    else             out1[(size_t)row * INNER + (col - INNER)] = siluf_(v);
                } else if (MODE == 1) {
                    out0[(size_t)row * N + col] = softplusf_(v + bias[col]);
                } else {
                    out0[(size_t)row * N + col] = v;
                }
            }
        }
    }
}

// ---------- depthwise conv(k=3, pad=1 over seq) + SiLU; emits f32 + bf16 hi/lo ----------
__global__ void k_conv_silu(const float* __restrict__ Inner,
                            const float* __restrict__ convw, const float* __restrict__ convb,
                            float* __restrict__ H,
                            uint16_t* __restrict__ Hhi, uint16_t* __restrict__ Hlo) {
    size_t tid = (size_t)blockIdx.x * blockDim.x + threadIdx.x;
    if (tid >= (size_t)MROWS * INNER) return;
    int i   = (int)(tid % INNER);
    int row = (int)(tid / INNER);
    int s   = row & (SEQ - 1);                     // seq pos within batch
    float cur  = Inner[tid];
    float prev = (s > 0)       ? Inner[tid - INNER] : 0.0f;
    float nxt  = (s < SEQ - 1) ? Inner[tid + INNER] : 0.0f;
    float c = prev * convw[i * 3 + 0] + cur * convw[i * 3 + 1] + nxt * convw[i * 3 + 2] + convb[i];
    float h = siluf_(c);
    H[tid] = h;
    uint16_t hh = f2bf(h);
    Hhi[tid] = hh;
    Hlo[tid] = f2bf(h - bf2f(hh));
}

// ---------- B-matrix projection: Bmat[row,n] = sum_k H[row,k] * W_state[k, 16+n] ----------
__global__ void k_bmat(const float* __restrict__ H, const float* __restrict__ Wstate,
                       float* __restrict__ Bmat) {
    int tid = blockIdx.x * blockDim.x + threadIdx.x;
    if (tid >= MROWS * NSTATE) return;
    int n   = tid & (NSTATE - 1);
    int row = tid >> 4;
    const float* h = H + (size_t)row * INNER;
    const float* w = Wstate + NSTATE + n;          // second chunk of [INNER, 2*NSTATE]
    float acc = 0.0f;
#pragma unroll 8
    for (int k = 0; k < INNER; ++k) acc += h[k] * w[(size_t)k * (2 * NSTATE)];
    Bmat[tid] = acc;
}

// ---------- chunked selective scan ----------
// thread = (b, chunk, i); maintains per-state vectors in registers
__global__ void k_scan_pass1(const float* __restrict__ dt, const float* __restrict__ H,
                             const float* __restrict__ Bmat, const float* __restrict__ A_log,
                             float* __restrict__ cProd, float* __restrict__ cComp) {
    int tid = blockIdx.x * blockDim.x + threadIdx.x;
    if (tid >= BATCH * NCHUNK * INNER) return;
    int i = tid % INNER;
    int c = (tid / INNER) % NCHUNK;
    int b = tid / (INNER * NCHUNK);

    float Arow[NSTATE], prod[NSTATE], comp[NSTATE];
#pragma unroll
    for (int n = 0; n < NSTATE; ++n) {
        Arow[n] = -__expf(A_log[i * NSTATE + n]);
        prod[n] = 1.0f;
        comp[n] = 0.0f;
    }
    int row0 = b * SEQ + c * LCHUNK;
    for (int t = 0; t < LCHUNK; ++t) {
        int row = row0 + t;
        float dtv = dt[(size_t)row * INNER + i];
        float hv  = H[(size_t)row * INNER + i];
        const float4* bp = reinterpret_cast<const float4*>(Bmat + (size_t)row * NSTATE);
        float Bv[NSTATE];
#pragma unroll
        for (int q = 0; q < 4; ++q) {
            float4 f = bp[q];
            Bv[4 * q + 0] = f.x; Bv[4 * q + 1] = f.y; Bv[4 * q + 2] = f.z; Bv[4 * q + 3] = f.w;
        }
        float dh = dtv * hv;
#pragma unroll
        for (int n = 0; n < NSTATE; ++n) {
            float e = __expf(dtv * Arow[n]);
            prod[n] *= e;
            comp[n] = e * comp[n] + dh * Bv[n];
        }
    }
    size_t o = (size_t)tid * NSTATE;               // ((b*NCHUNK+c)*INNER+i)*NSTATE
#pragma unroll
    for (int n = 0; n < NSTATE; ++n) { cProd[o + n] = prod[n]; cComp[o + n] = comp[n]; }
}

// sequential combine across the 16 chunks per (b,i,n)
__global__ void k_scan_pass2(const float* __restrict__ cProd, const float* __restrict__ cComp,
                             float* __restrict__ initState) {
    int tid = blockIdx.x * blockDim.x + threadIdx.x;
    if (tid >= BATCH * INNER * NSTATE) return;
    int n = tid % NSTATE;
    int i = (tid / NSTATE) % INNER;
    int b = tid / (NSTATE * INNER);
    float st = 0.0f;
    for (int c = 0; c < NCHUNK; ++c) {
        size_t o = ((size_t)(b * NCHUNK + c) * INNER + i) * NSTATE + n;
        initState[o] = st;
        st = cProd[o] * st + cComp[o];
    }
}

// replay chunk with correct incoming state; fuse y*silu(gate) and bf16 hi/lo split
__global__ void k_scan_pass3(const float* __restrict__ dt, const float* __restrict__ H,
                             const float* __restrict__ Bmat, const float* __restrict__ A_log,
                             const float* __restrict__ initState, const float* __restrict__ skip,
                             const float* __restrict__ Gate,
                             uint16_t* __restrict__ YgHi, uint16_t* __restrict__ YgLo) {
    int tid = blockIdx.x * blockDim.x + threadIdx.x;
    if (tid >= BATCH * NCHUNK * INNER) return;
    int i = tid % INNER;
    int c = (tid / INNER) % NCHUNK;
    int b = tid / (INNER * NCHUNK);

    float Arow[NSTATE], mem[NSTATE];
    size_t so = (size_t)tid * NSTATE;
#pragma unroll
    for (int n = 0; n < NSTATE; ++n) {
        Arow[n] = -__expf(A_log[i * NSTATE + n]);
        mem[n]  = initState[so + n];
    }
    float sk = skip[i];
    int row0 = b * SEQ + c * LCHUNK;
    for (int t = 0; t < LCHUNK; ++t) {
        int row = row0 + t;
        float dtv = dt[(size_t)row * INNER + i];
        float hv  = H[(size_t)row * INNER + i];
        const float4* bp = reinterpret_cast<const float4*>(Bmat + (size_t)row * NSTATE);
        float Bv[NSTATE];
#pragma unroll
        for (int q = 0; q < 4; ++q) {
            float4 f = bp[q];
            Bv[4 * q + 0] = f.x; Bv[4 * q + 1] = f.y; Bv[4 * q + 2] = f.z; Bv[4 * q + 3] = f.w;
        }
        float dh = dtv * hv;
        float y = 0.0f;
#pragma unroll
        for (int n = 0; n < NSTATE; ++n) {
            float e = __expf(dtv * Arow[n]);
            mem[n] = e * mem[n] + dh * Bv[n];
            y += mem[n];
        }
        y += sk * hv;
        size_t o = (size_t)row * INNER + i;
        float yg = y * Gate[o];
        uint16_t hh = f2bf(yg);
        YgHi[o] = hh;
        YgLo[o] = f2bf(yg - bf2f(hh));
    }
}

// ---------- host-side orchestration ----------
extern "C" void kernel_launch(void* const* d_in, const int* in_sizes, int n_in,
                              void* d_out, int out_size, void* d_ws, size_t ws_size,
                              hipStream_t stream) {
    (void)in_sizes; (void)n_in; (void)out_size; (void)ws_size;
    const float* x      = (const float*)d_in[0];
    const float* W_in   = (const float*)d_in[1];
    const float* conv_w = (const float*)d_in[2];
    const float* conv_b = (const float*)d_in[3];
    const float* W_state= (const float*)d_in[4];
    const float* W_time = (const float*)d_in[5];
    const float* b_time = (const float*)d_in[6];
    const float* A_log  = (const float*)d_in[7];
    const float* skip   = (const float*)d_in[8];
    const float* W_out  = (const float*)d_in[9];
    float* out = (float*)d_out;

    size_t off = 0;
    auto alloc = [&](size_t bytes) -> void* {
        void* p = (char*)d_ws + off;
        off += (bytes + 255) & ~(size_t)255;
        return p;
    };
    // bf16 split weights, transposed to [N,K]
    uint16_t* WinHiT = (uint16_t*)alloc((size_t)4096 * 1024 * 2);
    uint16_t* WinLoT = (uint16_t*)alloc((size_t)4096 * 1024 * 2);
    uint16_t* WtHiT  = (uint16_t*)alloc((size_t)2048 * 2048 * 2);
    uint16_t* WtLoT  = (uint16_t*)alloc((size_t)2048 * 2048 * 2);
    uint16_t* WoHiT  = (uint16_t*)alloc((size_t)1024 * 2048 * 2);
    uint16_t* WoLoT  = (uint16_t*)alloc((size_t)1024 * 2048 * 2);
    // bf16 split activations ([M,K] row-major)
    uint16_t* Xhi    = (uint16_t*)alloc((size_t)MROWS * DMODEL * 2);
    uint16_t* Xlo    = (uint16_t*)alloc((size_t)MROWS * DMODEL * 2);
    uint16_t* Hhi    = (uint16_t*)alloc((size_t)MROWS * INNER * 2);
    uint16_t* Hlo    = (uint16_t*)alloc((size_t)MROWS * INNER * 2);
    uint16_t* YgHi   = (uint16_t*)alloc((size_t)MROWS * INNER * 2);
    uint16_t* YgLo   = (uint16_t*)alloc((size_t)MROWS * INNER * 2);
    // fp32 intermediates
    float* Inner     = (float*)alloc((size_t)MROWS * INNER * 4);
    float* Gate      = (float*)alloc((size_t)MROWS * INNER * 4);
    float* H         = (float*)alloc((size_t)MROWS * INNER * 4);
    float* Dt        = (float*)alloc((size_t)MROWS * INNER * 4);
    float* Bm        = (float*)alloc((size_t)MROWS * NSTATE * 4);
    float* cProd     = (float*)alloc((size_t)BATCH * NCHUNK * INNER * NSTATE * 4);
    float* cComp     = (float*)alloc((size_t)BATCH * NCHUNK * INNER * NSTATE * 4);
    float* initState = (float*)alloc((size_t)BATCH * NCHUNK * INNER * NSTATE * 4);

    const int tpb = 256;
    auto nb = [](size_t total, int b) { return (unsigned)((total + b - 1) / b); };

    // precision-split + transpose weights, split x
    k_convert_weightT<<<nb((size_t)1024 * 4096, tpb), tpb, 0, stream>>>(W_in,  WinHiT, WinLoT, 1024, 4096);
    k_convert_weightT<<<nb((size_t)2048 * 2048, tpb), tpb, 0, stream>>>(W_time, WtHiT, WtLoT, 2048, 2048);
    k_convert_weightT<<<nb((size_t)2048 * 1024, tpb), tpb, 0, stream>>>(W_out,  WoHiT, WoLoT, 2048, 1024);
    k_convert_rowmajor<<<nb((size_t)MROWS * DMODEL, tpb), tpb, 0, stream>>>(x, Xhi, Xlo, (size_t)MROWS * DMODEL);

    // GEMM1: [2048,1024] @ [1024,4096] -> inner + silu(gate)
    k_gemm_bf16x3<0><<<dim3(4096 / 64, 2048 / 128), 128, 0, stream>>>(
        Xhi, Xlo, WinHiT, WinLoT, MROWS, 4096, 1024, Inner, Gate, nullptr);

    // conv + silu -> H (f32 + bf16 split)
    k_conv_silu<<<nb((size_t)MROWS * INNER, tpb), tpb, 0, stream>>>(Inner, conv_w, conv_b, H, Hhi, Hlo);

    // B-matrix projection (N=32 too small for WMMA)
    k_bmat<<<nb((size_t)MROWS * NSTATE, tpb), tpb, 0, stream>>>(H, W_state, Bm);

    // GEMM2: [2048,2048] @ [2048,2048] + bias -> softplus -> dt
    k_gemm_bf16x3<1><<<dim3(2048 / 64, 2048 / 128), 128, 0, stream>>>(
        Hhi, Hlo, WtHiT, WtLoT, MROWS, 2048, 2048, Dt, nullptr, b_time);

    // chunked selective scan
    k_scan_pass1<<<nb((size_t)BATCH * NCHUNK * INNER, tpb), tpb, 0, stream>>>(Dt, H, Bm, A_log, cProd, cComp);
    k_scan_pass2<<<nb((size_t)BATCH * INNER * NSTATE, tpb), tpb, 0, stream>>>(cProd, cComp, initState);
    k_scan_pass3<<<nb((size_t)BATCH * NCHUNK * INNER, tpb), tpb, 0, stream>>>(
        Dt, H, Bm, A_log, initState, skip, Gate, YgHi, YgLo);

    // GEMM3: [2048,2048] @ [2048,1024] -> out
    k_gemm_bf16x3<2><<<dim3(1024 / 64, 2048 / 128), 128, 0, stream>>>(
        YgHi, YgLo, WoHiT, WoLoT, MROWS, 1024, 2048, out, nullptr, nullptr);
}
